// ChamferLoss_51814485459453
// MI455X (gfx1250) — compile-verified, matrix operand order checked by reading, not run on hardware
//
#include <hip/hip_runtime.h>

// ---- CDNA5 (gfx1250) wave32 WMMA types ----
typedef __attribute__((ext_vector_type(16))) __bf16 v16bf;
typedef __attribute__((ext_vector_type(8)))  float  v8f;
typedef __attribute__((ext_vector_type(4))) unsigned int uint4v;
typedef __attribute__((ext_vector_type(4)))  float  floatx4;

#define BQ 16
#define NQ 4096
#define MQ 4096
#define DQ 64

#define TM 64              // X rows per workgroup
#define TN 64              // Y rows per j-step
#define KST 72             // padded bf16 row stride in LDS (144B = 9 uint4, conflict-free)

// LDS layout (bytes), one flat aliased block:
//   [0      , 18432) : Y tile buffer 0 (hi @ +0, lo @ +9216)
//   [18432  , 36864) : Y tile buffer 1 -- also X staging region (dead after A-frag build)
//   [36864  , 37888) : x2 partials (256 floats)
//   [37888  , 54272) : running column mins (4096 uint)
#define REGION  18432
#define LO_OFF   9216
#define X2_OFF  36864
#define CM_OFF  37888
#define SMEM_BYTES 54272

union FragU { uint4v u[2]; v16bf v; };

__device__ __forceinline__ unsigned f32_to_bf16_rne(float f) {
    unsigned u = __float_as_uint(f);
    unsigned r = u + 0x7FFFu + ((u >> 16) & 1u);
    return r >> 16;
}

#define WMMA_BF16(A, Bv, C) \
    __builtin_amdgcn_wmma_f32_16x16x32_bf16(false, (A), false, (Bv), (short)0, (C), false, false)

// min with a ds_swizzle lane-exchange (group-of-32 xor pattern, immediate offset)
#define SWZ_MIN(v, imm) \
    (v) = fminf((v), __int_as_float(__builtin_amdgcn_ds_swizzle(__float_as_int(v), (imm))))

__global__ void init_mins_kernel(unsigned* __restrict__ p, int n) {
    int i = blockIdx.x * blockDim.x + threadIdx.x;
    if (i < n) p[i] = 0x7F800000u;   // +inf bits
}

// fp32 [nrows,64] -> planar bf16 hi/lo [nrows,64] + |row|^2 (exact fp32).
// 4 threads per row, fully coalesced b128 traffic. Launched with exactly nrows*4 threads.
__global__ void __launch_bounds__(256)
convert_bf16hilo_kernel(const float* __restrict__ src, unsigned short* __restrict__ hiP,
                        unsigned short* __restrict__ loP, float* __restrict__ sq)
{
    int gt  = blockIdx.x * 256 + threadIdx.x;
    int row = gt >> 2;
    int ks  = (gt & 3) * 16;
    const floatx4* s4 = (const floatx4*)(src + (size_t)row * DQ + ks);

    float fv[16];
    #pragma unroll
    for (int q = 0; q < 4; ++q) {
        floatx4 v = s4[q];
        fv[q*4+0] = v[0]; fv[q*4+1] = v[1]; fv[q*4+2] = v[2]; fv[q*4+3] = v[3];
    }
    unsigned hw[8], lw[8];
    float s = 0.f;
    #pragma unroll
    for (int p = 0; p < 8; ++p) {
        float f0 = fv[2*p], f1 = fv[2*p+1];
        s += f0 * f0;
        s += f1 * f1;
        unsigned h0 = f32_to_bf16_rne(f0), h1 = f32_to_bf16_rne(f1);
        float hf0 = __uint_as_float(h0 << 16), hf1 = __uint_as_float(h1 << 16);
        unsigned l0 = f32_to_bf16_rne(f0 - hf0), l1 = f32_to_bf16_rne(f1 - hf1);
        hw[p] = h0 | (h1 << 16);
        lw[p] = l0 | (l1 << 16);
    }
    uint4v* hd = (uint4v*)(hiP + (size_t)row * DQ + ks);
    uint4v* ld = (uint4v*)(loP + (size_t)row * DQ + ks);
    uint4v hv0 = {hw[0], hw[1], hw[2], hw[3]}, hv1 = {hw[4], hw[5], hw[6], hw[7]};
    uint4v lv0 = {lw[0], lw[1], lw[2], lw[3]}, lv1 = {lw[4], lw[5], lw[6], lw[7]};
    hd[0] = hv0; hd[1] = hv1;
    ld[0] = lv0; ld[1] = lv1;

    // reduce |row|^2 across the 4 lanes of the quad
    s += __int_as_float(__builtin_amdgcn_ds_swizzle(__float_as_int(s), 0x041F)); // SWAPX1
    s += __int_as_float(__builtin_amdgcn_ds_swizzle(__float_as_int(s), 0x081F)); // SWAPX2
    if ((gt & 3) == 0) sq[row] = s;
}

__global__ void __launch_bounds__(256)
chamfer_tiles_kernel(const float* __restrict__ X,
                     const unsigned short* __restrict__ YhiG,
                     const unsigned short* __restrict__ YloG,
                     const float* __restrict__ y2G,
                     unsigned* __restrict__ rowminG, unsigned* __restrict__ colminG)
{
    __shared__ uint4v smem4[SMEM_BYTES / 16];
    unsigned char* smem = (unsigned char*)smem4;
    float*    x2p     = (float*)(smem + X2_OFF);
    unsigned* colminL = (unsigned*)(smem + CM_OFF);

    const int b    = blockIdx.y;
    const int i0   = blockIdx.x * TM;
    const int t    = threadIdx.x;
    const int lane = t & 31;
    const int wave = t >> 5;
    const int idx  = lane & 15;           // row/col index inside a 16x16 tile
    const int h    = lane >> 4;           // lane half selects k-subgroup / row half

    const float* Xb = X + ((size_t)b * NQ + i0) * DQ;
    const uint4v* YhiB = (const uint4v*)(YhiG + (size_t)b * MQ * DQ);
    const uint4v* YloB = (const uint4v*)(YloG + (size_t)b * MQ * DQ);

    // init LDS column mins to +inf
    #pragma unroll
    for (int q = 0; q < MQ / 256; ++q) colminL[q * 256 + t] = 0x7F800000u;

    // ---- stage X tile (into Y buffer 1 region) as bf16 hi/lo + |x|^2 partials ----
    {
        unsigned short* XhiS = (unsigned short*)(smem + REGION);
        unsigned short* XloS = (unsigned short*)(smem + REGION + LO_OFF);
        int row = t >> 2;
        int ks  = (t & 3) * 16;
        const floatx4* src = (const floatx4*)(Xb + row * DQ + ks);
        float s = 0.f;
        #pragma unroll
        for (int q = 0; q < 4; ++q) {
            floatx4 v = src[q];
            #pragma unroll
            for (int e = 0; e < 4; ++e) {
                float f = v[e];
                s += f * f;
                unsigned hb = f32_to_bf16_rne(f);
                float hf = __uint_as_float(hb << 16);
                unsigned lb = f32_to_bf16_rne(f - hf);
                int o = row * KST + ks + q * 4 + e;
                XhiS[o] = (unsigned short)hb; XloS[o] = (unsigned short)lb;
            }
        }
        x2p[t] = s;
    }

    // ---- per-wave tile assignment: 32 rows (2 row tiles) x 16 cols ----
    const int rowbase = (wave >> 2) * 32;        // 0 or 32
    const int colbase = (wave & 3) * 16;         // 0,16,32,48

    // preload first Y tile (pure copy; pre-converted bf16 planes) + first y^2
    uint4v nph[2], npl[2];
    #pragma unroll
    for (int p = 0; p < 2; ++p) {
        int g = 2 * t + p;                 // uint4 index in [0, 512)
        nph[p] = YhiB[g];
        npl[p] = YloB[g];
    }
    float y2cur = y2G[b * MQ + colbase + idx];
    __syncthreads();                       // X tile staged

    // A fragments for both row tiles (16x32 bf16 per k-chunk), resident in VGPRs
    FragU Ahi[2][2], Alo[2][2];            // [row tile][k chunk]
    {
        const uint4v* xh = (const uint4v*)(smem + REGION);
        const uint4v* xl = (const uint4v*)(smem + REGION + LO_OFF);
        #pragma unroll
        for (int rt = 0; rt < 2; ++rt) {
            int row = rowbase + rt * 16 + idx;
            #pragma unroll
            for (int c = 0; c < 2; ++c) {
                int base = row * 9 + c * 4 + h;      // (row*144 + c*64 + h*16)/16
                Ahi[rt][c].u[0] = xh[base];  Ahi[rt][c].u[1] = xh[base + 2];
                Alo[rt][c].u[0] = xl[base];  Alo[rt][c].u[1] = xl[base + 2];
            }
        }
    }
    // per-lane x^2 for the 8 rows each accumulator covers (M = rowbase + rt*16 + h*8 + r)
    float x2r[2][8];
    float rmin[2][8];
    #pragma unroll
    for (int rt = 0; rt < 2; ++rt)
        #pragma unroll
        for (int r = 0; r < 8; ++r) {
            int row = rowbase + rt * 16 + h * 8 + r;
            x2r[rt][r] = x2p[row * 4 + 0] + x2p[row * 4 + 1] + x2p[row * 4 + 2] + x2p[row * 4 + 3];
            rmin[rt][r] = 3.4e38f;
        }

    // store first Y tile into buffer 0 (padded rows: uint4 index row*9 + c)
    {
        uint4v* lh = (uint4v*)(smem);
        uint4v* ll = (uint4v*)(smem + LO_OFF);
        #pragma unroll
        for (int p = 0; p < 2; ++p) {
            int g = 2 * t + p;
            int o = (g >> 3) * 9 + (g & 7);
            lh[o] = nph[p]; ll[o] = npl[p];
        }
    }
    __syncthreads();                       // tile 0 visible; all A-frag reads of buffer 1 done

    // ---- sweep all Y tiles: zero-cost double buffering, ONE barrier per iteration ----
    int pb = 0;
    #pragma unroll 2
    for (int j0 = 0; j0 < MQ; j0 += TN) {
        const bool have_next = (j0 + TN < MQ);
        float y2nxt = y2cur;
        if (have_next) {                   // issue next tile's global loads early
            int gbase = (j0 + TN) * 8;     // 8 uint4 per row
            #pragma unroll
            for (int p = 0; p < 2; ++p) {
                int g = 2 * t + p;
                nph[p] = YhiB[gbase + g];
                npl[p] = YloB[gbase + g];
            }
            y2nxt = y2G[b * MQ + j0 + TN + colbase + idx];
        }

        // B fragments for this wave's single 16-col tile (shared by both row tiles)
        FragU Bhi[2], Blo[2];              // [k chunk]
        {
            const uint4v* yh = (const uint4v*)(smem + pb * REGION);
            const uint4v* yl = (const uint4v*)(smem + pb * REGION + LO_OFF);
            int col = colbase + idx;
            #pragma unroll
            for (int c = 0; c < 2; ++c) {
                int base = col * 9 + c * 4 + h;
                Bhi[c].u[0] = yh[base];  Bhi[c].u[1] = yh[base + 2];
                Blo[c].u[0] = yl[base];  Blo[c].u[1] = yl[base + 2];
            }
        }

        // bf16x3 fp32-emulated xy, two interleaved accumulator chains (one per row tile)
        v8f acc[2] = {{}, {}};
        #pragma unroll
        for (int c = 0; c < 2; ++c) {
            #pragma unroll
            for (int rt = 0; rt < 2; ++rt) acc[rt] = WMMA_BF16(Ahi[rt][c].v, Blo[c].v, acc[rt]);
            #pragma unroll
            for (int rt = 0; rt < 2; ++rt) acc[rt] = WMMA_BF16(Alo[rt][c].v, Bhi[c].v, acc[rt]);
            #pragma unroll
            for (int rt = 0; rt < 2; ++rt) acc[rt] = WMMA_BF16(Ahi[rt][c].v, Bhi[c].v, acc[rt]);
        }

        // epilogue: distances + fused min reductions (both row tiles share one column)
        float cmin = 3.4e38f;
        #pragma unroll
        for (int rt = 0; rt < 2; ++rt) {
            #pragma unroll
            for (int r = 0; r < 8; ++r) {
                float d = fmaxf(x2r[rt][r] + y2cur - 2.0f * acc[rt][r], 0.0f);
                rmin[rt][r] = fminf(rmin[rt][r], d);
                cmin = fminf(cmin, d);
            }
        }
        // lanes L and L^16 hit the same address (both row halves) -> ds_min_u32
        atomicMin(&colminL[j0 + colbase + idx], __float_as_uint(cmin));
        y2cur = y2nxt;

        // store prefetched tile into the buffer nobody reads this iteration
        if (have_next) {
            uint4v* lh = (uint4v*)(smem + (pb ^ 1) * REGION);
            uint4v* ll = (uint4v*)(smem + (pb ^ 1) * REGION + LO_OFF);
            #pragma unroll
            for (int p = 0; p < 2; ++p) {
                int g = 2 * t + p;
                int o = (g >> 3) * 9 + (g & 7);
                lh[o] = nph[p]; ll[o] = npl[p];
            }
        }
        __syncthreads();                   // next tile visible; current buffer free after next iter
        pb ^= 1;
    }

    // ---- row mins: reduce over the 16 lanes of each half via ds_swizzle, then global ----
    #pragma unroll
    for (int rt = 0; rt < 2; ++rt)
        #pragma unroll
        for (int r = 0; r < 8; ++r) {
            SWZ_MIN(rmin[rt][r], 0x041F);  // SWAPX1
            SWZ_MIN(rmin[rt][r], 0x081F);  // SWAPX2
            SWZ_MIN(rmin[rt][r], 0x101F);  // SWAPX4
            SWZ_MIN(rmin[rt][r], 0x201F);  // SWAPX8
        }
    if (idx == 0) {
        #pragma unroll
        for (int rt = 0; rt < 2; ++rt)
            #pragma unroll
            for (int r = 0; r < 8; ++r) {
                int row = i0 + rowbase + rt * 16 + h * 8 + r;
                atomicMin(&rowminG[b * NQ + row], __float_as_uint(rmin[rt][r]));
            }
    }

    // ---- flush column mins once per workgroup ----
    __syncthreads();
    #pragma unroll
    for (int q = 0; q < MQ / 256; ++q) {
        int j = q * 256 + t;
        atomicMin(&colminG[b * MQ + j], colminL[j]);
    }
}

__global__ void __launch_bounds__(256)
reduce_sum_kernel(const unsigned* __restrict__ rowminG,
                  const unsigned* __restrict__ colminG, float* __restrict__ out)
{
    __shared__ float sh[256];
    int t = threadIdx.x;
    float s = 0.f;
    const int total = BQ * NQ;
    for (int i = t; i < total; i += 256) s += __uint_as_float(rowminG[i]);
    for (int i = t; i < total; i += 256) s += __uint_as_float(colminG[i]);
    sh[t] = s;
    __syncthreads();
    for (int w = 128; w > 0; w >>= 1) {
        if (t < w) sh[t] += sh[t + w];
        __syncthreads();
    }
    if (t == 0) out[0] = 0.5f * sh[0];
}

extern "C" void kernel_launch(void* const* d_in, const int* in_sizes, int n_in,
                              void* d_out, int out_size, void* d_ws, size_t ws_size,
                              hipStream_t stream) {
    const float* X = (const float*)d_in[0];   // [B, N, D] fp32
    const float* Y = (const float*)d_in[1];   // [B, M, D] fp32
    float* out = (float*)d_out;

    // workspace layout (all 16B-aligned):
    unsigned* rowminG = (unsigned*)d_ws;                         // B*N
    unsigned* colminG = rowminG + (size_t)BQ * NQ;               // B*M
    float*    y2G     = (float*)(colminG + (size_t)BQ * MQ);     // B*M
    unsigned short* YhiG = (unsigned short*)(y2G + (size_t)BQ * MQ);   // B*M*D
    unsigned short* YloG = YhiG + (size_t)BQ * MQ * DQ;                // B*M*D

    int total = BQ * NQ + BQ * MQ;
    init_mins_kernel<<<(total + 255) / 256, 256, 0, stream>>>(rowminG, total);

    // one-time fp32 -> bf16 hi/lo pre-conversion of the streamed operand (memory-bound)
    int yrows = BQ * MQ;                    // 65536 rows, 4 threads each
    convert_bf16hilo_kernel<<<(yrows * 4) / 256, 256, 0, stream>>>(Y, YhiG, YloG, y2G);

    dim3 grid(NQ / TM, BQ);
    chamfer_tiles_kernel<<<grid, 256, 0, stream>>>(X, YhiG, YloG, y2G, rowminG, colminG);

    reduce_sum_kernel<<<1, 256, 0, stream>>>(rowminG, colminG, out);
}
